// Ultimate_STCN_GraphSNN_35742717837383
// MI455X (gfx1250) — matrix operand order repaired and checked by reading, not compiled, not run on hardware
//
#include <hip/hip_runtime.h>
#include <hip/hip_bf16.h>
#include <math.h>

typedef __attribute__((ext_vector_type(16))) _Float16 v16h;
typedef __attribute__((ext_vector_type(8)))  float    v8f;

#define WMMA_F16(a,b,c) __builtin_amdgcn_wmma_f32_16x16x32_f16(false,(a),false,(b),(short)0,(c),false,false)

union Frag32B { uint4 u[2]; v16h h; };

// Load a v16h fragment as two 16-byte chunks (both must be 16B aligned).
__device__ __forceinline__ v16h ld_frag(const _Float16* p0, const _Float16* p1) {
  Frag32B f;
  f.u[0] = *(const uint4*)p0;
  f.u[1] = *(const uint4*)p1;
  return f.h;
}

// ---- pre-kernel: reorder attn_w1 [2048][32] f32 into WMMA B-fragment-native f16 ----
// layout: idx = ((kc*2 + nt)*32 + lane)*16 + hh  ;  B elem = attn_w1[32*kc + 16*(lane>=16) + hh][16*nt + (lane&15)]
__global__ void reorder_attnw1_k(const float* __restrict__ w1, _Float16* __restrict__ outp) {
  int idx = blockIdx.x * 256 + threadIdx.x;          // 65536 total
  int hh   = idx & 15;
  int lane = (idx >> 4) & 31;
  int nt   = (idx >> 9) & 1;
  int kc   = idx >> 10;
  int row  = 32*kc + ((lane >= 16) ? 16 : 0) + hh;
  int col  = 16*nt + (lane & 15);
  outp[idx] = (_Float16)w1[row*32 + col];
}

#define SMEM_BYTES 118848

__global__ __launch_bounds__(128)
void stcn_fused_k(const float* __restrict__ Lnorm, const float* __restrict__ Xseq,
                  const float* __restrict__ conv_w, const float* __restrict__ conv_b,
                  const float* __restrict__ bn_g, const float* __restrict__ bn_b,
                  const float* __restrict__ bn_m, const float* __restrict__ bn_v,
                  const float* __restrict__ g1w, const float* __restrict__ g1b,
                  const float* __restrict__ mask1,
                  const float* __restrict__ g2w, const float* __restrict__ g2b,
                  const float* __restrict__ mask2,
                  const float* __restrict__ beta2p,
                  const float* __restrict__ attn_w1, const float* __restrict__ attn_b1,
                  const float* __restrict__ attn_w2, const float* __restrict__ attn_b2,
                  const float* __restrict__ cls_w1, const float* __restrict__ cls_b1,
                  const float* __restrict__ ln_g, const float* __restrict__ ln_b,
                  const float* __restrict__ cls_w2, const float* __restrict__ cls_b2,
                  const _Float16* __restrict__ w1f,   // may be null -> slow gather path
                  float* __restrict__ outp)
{
  extern __shared__ char smem[];
  float*    sL1   = (float*)(smem + 0);        // [64][64] f32 L*m1
  _Float16* sL2h  = (_Float16*)(smem + 16384); // [64][64] f16 L*m2
  float*    sX    = (float*)(smem + 24576);    // [64][16] input (col t)
  float*    sY    = (float*)(smem + 28672);    // [64][16] conv+BN out
  float*    sSpkT = (float*)(smem + 32768);    // [64]
  float*    sV    = (float*)(smem + 33024);    // [64]  L1 @ spk
  float*    sSpk1 = (float*)(smem + 33280);    // [64][16]
  _Float16* sA2T  = (_Float16*)(smem + 37376); // [32][72] transposed A2 (f,j)
  float*    sW2   = (float*)(smem + 41984);    // [16][32]
  float*    sW1   = (float*)(smem + 44032);    // [16]
  float*    sB1   = (float*)(smem + 44096);    // [16]
  float*    sB2   = (float*)(smem + 44160);    // [32]
  float*    sScr  = (float*)(smem + 44288);    // [16] attn scores
  float*    sH    = (float*)(smem + 44352);    // [128]
  float*    sCtx  = (float*)(smem + 44864);    // [2048] (also attn partial scratch)
  _Float16* sPots = (_Float16*)(smem + 53056); // [16][2056] f16 mem_g2 history

  const int tid  = threadIdx.x;
  const int lane = tid & 31;
  const int wv   = tid >> 5;
  const int b    = blockIdx.x;
  const float beta2 = beta2p[0];

  // prefetch conv_w (L2/WGP$) while the sigmoid/LDS-fill VALU work runs
  for (int i = tid * 64; i < 12288; i += 128 * 64)
    __builtin_prefetch(conv_w + i, 0, 3);

  // ---- masks, L split, small weights, X ----
  for (int i = tid; i < 4096; i += 128) {
    int r = i >> 6, c = i & 63;
    float mm1 = mask1[i] + mask1[c*64 + r];
    float mm2 = mask2[i] + mask2[c*64 + r];
    float m1 = 0.5f / (1.f + __expf(-mm1));
    float m2 = 0.5f / (1.f + __expf(-mm2));
    float L = Lnorm[b*4096 + i];
    sL1[i]  = L * m1;
    sL2h[i] = (_Float16)(L * m2);
  }
  for (int i = tid; i < 512; i += 128) sW2[i] = g2w[i];
  if (tid < 16) sW1[tid] = g1w[tid];
  if (tid < 16) sB1[tid] = g1b[tid];
  if (tid < 32) sB2[tid] = g2b[tid];
  for (int i = tid; i < 960; i += 128) {        // X_seq[b][t][n][0] -> sX[n][t]
    int t = i >> 6, n = i & 63;
    sX[n*16 + t] = Xseq[b*960 + i];
  }
  __syncthreads();

  // ---- causal conv1d (k=3, pad-left 2) + inference BN ----
  for (int i = tid; i < 960; i += 128) {
    int c = i / 15, t = i - c*15;
    float acc = conv_b[c];
    const float* wr = conv_w + c*192;
#pragma unroll 4
    for (int ic = 0; ic < 64; ++ic) {
      float x0 = (t >= 2) ? sX[ic*16 + t-2] : 0.f;
      float x1 = (t >= 1) ? sX[ic*16 + t-1] : 0.f;
      float x2 = sX[ic*16 + t];
      acc = fmaf(wr[ic*3+0], x0, acc);
      acc = fmaf(wr[ic*3+1], x1, acc);
      acc = fmaf(wr[ic*3+2], x2, acc);
    }
    float g = bn_g[c] * rsqrtf(bn_v[c] + 1e-5f);
    sY[c*16 + t] = (acc - bn_m[c]) * g + bn_b[c];
  }
  __syncthreads();

  // ---- L2 A-fragments, built once, reused across all 15 steps ----
  // A layout (16-bit 16x32): lane<16 -> k = {kc*32 + 0..7, kc*32+16..23}; lane>=16 -> +8
  v16h la0, la1;
  {
    int m   = (wv << 4) + (lane & 15);
    int kb0 = (lane >> 4) ? 8 : 0;
    const _Float16* row = sL2h + m*64;
    la0 = ld_frag(row + kb0,      row + kb0 + 16);
    la1 = ld_frag(row + 32 + kb0, row + 32 + kb0 + 16);
  }

  // ---- LIF states in registers ----
  float mem_t = 0.f;
  float mem1[8];
#pragma unroll
  for (int j = 0; j < 8; ++j) mem1[j] = 0.f;
  float mem2a[2][8];
#pragma unroll
  for (int n = 0; n < 2; ++n)
#pragma unroll
    for (int r = 0; r < 8; ++r) mem2a[n][r] = 0.f;

  const int n1 = tid >> 1;            // node for layer-1 state (2 threads/node)
  const int q0 = (tid & 1) * 8;
  const int jA = tid >> 1;            // node for A2 compute
  const int fA = (tid & 1) * 16;

  // ================= time recurrence =================
  for (int t = 0; t < 15; ++t) {
    // layer-0 LIF (per-node scalar)
    if (tid < 64) {
      float reset = (mem_t - 1.f > 0.f) ? 1.f : 0.f;
      mem_t = 0.9f*mem_t + sY[tid*16 + t] - reset;
      sSpkT[tid] = (mem_t - 1.f > 0.f) ? 1.f : 0.f;
    }
    __syncthreads();
    // v = L1 @ spk_t  (rank-1 trick: cur1 = v ⊗ gcn1_w)
    if (tid < 64) {
      const float* Lr = sL1 + tid*64;
      float acc = 0.f;
#pragma unroll 8
      for (int j = 0; j < 64; ++j) acc = fmaf(Lr[j], sSpkT[j], acc);
      sV[tid] = acc;
    }
    __syncthreads();
    // layer-1 LIF
    {
      float v = sV[n1];
#pragma unroll
      for (int j = 0; j < 8; ++j) {
        int q = q0 + j;
        float cur = fmaf(v, sW1[q], sB1[q]);
        float reset = (mem1[j] - 1.f > 0.f) ? 1.f : 0.f;
        mem1[j] = 0.85f*mem1[j] + cur - reset;
        sSpk1[n1*16 + q] = (mem1[j] - 1.f > 0.f) ? 1.f : 0.f;
      }
    }
    __syncthreads();
    // A2 = spk1 @ gcn2_w, stored transposed (f-major) for B-fragment loads
    for (int f = fA; f < fA + 16; ++f) {
      float acc = 0.f;
#pragma unroll
      for (int q = 0; q < 16; ++q)
        acc = fmaf(sSpk1[jA*16 + q], sW2[q*32 + f], acc);
      sA2T[f*72 + jA] = (_Float16)acc;
    }
    __syncthreads();
    // cur2 = L2 @ A2 via WMMA; layer-2 LIF; record mem_g2 into pots (f16)
    {
      int c = lane & 15;
      int kofs = (lane >> 4) ? 8 : 0;           // B layout: lane>=16 holds k=8..15,24..31 of chunk
      int mbase = (wv << 4) + ((lane >> 4) << 3);
#pragma unroll
      for (int nt = 0; nt < 2; ++nt) {
        const _Float16* colp = sA2T + (nt*16 + c)*72 + kofs;
        v16h bf0 = ld_frag(colp,      colp + 16);
        v16h bf1 = ld_frag(colp + 32, colp + 48);
        v8f acc = {};
        acc = WMMA_F16(la0, bf0, acc);
        acc = WMMA_F16(la1, bf1, acc);
        int f = nt*16 + c;
        float bias = sB2[f];
#pragma unroll
        for (int r = 0; r < 8; ++r) {
          float cur = acc[r] + bias;
          float m = mem2a[nt][r];
          float reset = (m - 1.f > 0.f) ? 1.f : 0.f;
          m = beta2*m + cur - reset;
          mem2a[nt][r] = m;
          sPots[t*2056 + (mbase + r)*32 + f] = (_Float16)m;
        }
      }
    }
    __syncthreads();
  }

  // pad row t=15 with zeros (WMMA M=16), init score accumulators
  for (int i = tid; i < 2056; i += 128) sPots[15*2056 + i] = (_Float16)0.f;
  if (tid < 16) sScr[tid] = attn_b2[0];
  __syncthreads();

  // ---- attention scores: (pots 16x2048) @ (attn_w1 2048x32) via WMMA ----
  // All 4 waves: wave = {nt, khalf}. Waves 2/3 spill partial tiles through LDS,
  // waves 0/1 combine BEFORE the tanh nonlinearity.
  {
    const int nt = wv & 1;
    const int kh = wv >> 1;
    const int c  = lane & 15;
    v8f acc = {};
    for (int kc = kh*32; kc < kh*32 + 32; ++kc) {
      int kb = kc*32 + ((lane >> 4) ? 16 : 0);
      const _Float16* rowp = sPots + c*2056 + kb;     // row = t = lane&15
      v16h af = ld_frag(rowp, rowp + 8);
      v16h bf;
      if (w1f) {
        const _Float16* gp = w1f + (((kc*2 + nt)*32 + lane) << 4);
        bf = ld_frag(gp, gp + 8);
      } else {
        int kbase = (lane >> 4) ? 16 : 0;
        int col = nt*16 + c;
#pragma unroll
        for (int hh = 0; hh < 16; ++hh)
          bf[hh] = (_Float16)attn_w1[(kc*32 + kbase + hh)*32 + col];
      }
      acc = WMMA_F16(af, bf, acc);
    }
    const int pbase = nt*256 + ((lane >> 4) ? 128 : 0);
    if (wv >= 2) {            // wave-uniform branch; EXEC stays full in-wave
#pragma unroll
      for (int r = 0; r < 8; ++r)
        sCtx[pbase + r*16 + c] = acc[r];
    }
    __syncthreads();
    if (wv < 2) {
      int f = nt*16 + c;
      float ab1 = attn_b1[f], aw2 = attn_w2[f];
#pragma unroll
      for (int r = 0; r < 8; ++r) {
        int tt = r + ((lane >> 4) << 3);
        float s = acc[r] + sCtx[pbase + r*16 + c];    // combine K-halves
        atomicAdd(&sScr[tt], tanhf(s + ab1) * aw2);
      }
    }
  }
  __syncthreads();

  // ---- softmax over T=15 ----
  if (tid == 0) {
    float mx = sScr[0];
    for (int t = 1; t < 15; ++t) mx = fmaxf(mx, sScr[t]);
    float s = 0.f;
    for (int t = 0; t < 15; ++t) { float e = __expf(sScr[t] - mx); sScr[t] = e; s += e; }
    float inv = 1.f / s;
    for (int t = 0; t < 15; ++t) sScr[t] *= inv;
  }
  __syncthreads();

  // ---- ctx = sum_t w[t] * pots[t] ----
  for (int k = tid; k < 2048; k += 128) {
    float acc = 0.f;
#pragma unroll
    for (int t = 0; t < 15; ++t)
      acc = fmaf(sScr[t], (float)sPots[t*2056 + k], acc);
    sCtx[k] = acc;
  }
  __syncthreads();

  // ---- h = ctx @ cls_w1 + b (coalesced: lane tid reads column tid) ----
  {
    float acc = cls_b1[tid];
    for (int k = 0; k < 2048; ++k)
      acc = fmaf(sCtx[k], cls_w1[k*128 + tid], acc);
    sH[tid] = acc;
  }
  __syncthreads();

  // ---- LayerNorm + exact GELU ----
  {
    float s = 0.f, s2 = 0.f;
#pragma unroll 8
    for (int i = 0; i < 128; ++i) { float x = sH[i]; s += x; s2 += x*x; }
    float mu  = s * (1.f/128.f);
    float var = s2 * (1.f/128.f) - mu*mu;
    float x = (sH[tid] - mu) * rsqrtf(var + 1e-5f) * ln_g[tid] + ln_b[tid];
    float ge = 0.5f * x * (1.f + erff(x * 0.70710678f));
    __syncthreads();
    sH[tid] = ge;
  }
  __syncthreads();

  // ---- out = h @ cls_w2 + b2 ----
  if (tid < 4) {
    float acc = cls_b2[tid];
#pragma unroll 8
    for (int i = 0; i < 128; ++i)
      acc = fmaf(sH[i], cls_w2[i*4 + tid], acc);
    outp[b*4 + tid] = acc;
  }
}

extern "C" void kernel_launch(void* const* d_in, const int* in_sizes, int n_in,
                              void* d_out, int out_size, void* d_ws, size_t ws_size,
                              hipStream_t stream) {
  const float* Lnorm   = (const float*)d_in[0];
  const float* Xseq    = (const float*)d_in[1];
  const float* conv_w  = (const float*)d_in[2];
  const float* conv_b  = (const float*)d_in[3];
  const float* bn_g    = (const float*)d_in[4];
  const float* bn_b    = (const float*)d_in[5];
  const float* bn_m    = (const float*)d_in[6];
  const float* bn_v    = (const float*)d_in[7];
  const float* g1w     = (const float*)d_in[8];
  const float* g1b     = (const float*)d_in[9];
  const float* mask1   = (const float*)d_in[10];
  const float* g2w     = (const float*)d_in[11];
  const float* g2b     = (const float*)d_in[12];
  const float* mask2   = (const float*)d_in[13];
  const float* beta2p  = (const float*)d_in[14];
  const float* attn_w1 = (const float*)d_in[15];
  const float* attn_b1 = (const float*)d_in[16];
  const float* attn_w2 = (const float*)d_in[17];
  const float* attn_b2 = (const float*)d_in[18];
  const float* cls_w1  = (const float*)d_in[19];
  const float* cls_b1  = (const float*)d_in[20];
  const float* ln_g    = (const float*)d_in[21];
  const float* ln_b    = (const float*)d_in[22];
  const float* cls_w2  = (const float*)d_in[23];
  const float* cls_b2  = (const float*)d_in[24];

  int B = in_sizes[1] / 960;     // X_seq: B*15*64

  _Float16* w1f = nullptr;
  if (ws_size >= 65536 * sizeof(_Float16)) {
    w1f = (_Float16*)d_ws;
    reorder_attnw1_k<<<256, 256, 0, stream>>>(attn_w1, w1f);
  }

  stcn_fused_k<<<B, 128, SMEM_BYTES, stream>>>(
      Lnorm, Xseq, conv_w, conv_b, bn_g, bn_b, bn_m, bn_v,
      g1w, g1b, mask1, g2w, g2b, mask2, beta2p,
      attn_w1, attn_b1, attn_w2, attn_b2,
      cls_w1, cls_b1, ln_g, ln_b, cls_w2, cls_b2,
      w1f, (float*)d_out);
}
